// GraphAttentionLayer_7507602833556
// MI455X (gfx1250) — matrix operand order repaired and checked by reading, not compiled, not run on hardware
//
#include <hip/hip_runtime.h>

typedef __attribute__((ext_vector_type(16))) _Float16 v16h;
typedef __attribute__((ext_vector_type(8)))  _Float16 v8h;
typedef __attribute__((ext_vector_type(8)))  float    v8f;

#define GN     8192
#define IN_F   64
#define OUT_F  64
#define D_INT  32
#define ALPHA  0.2f
#define KSPLIT 4
#define KSEG   (GN / KSPLIT)

__device__ __forceinline__ float lrelu(float v) { return fmaxf(v, ALPHA * v); }
__device__ __forceinline__ float elu_f(float v) { return v > 0.f ? v : (__expf(v) - 1.f); }

// ---------------------------------------------------------------------------
// K1: h = x @ W ; s = h@a_src + intent@b_src ; t = h@a_dst + intent@b_dst
// ---------------------------------------------------------------------------
__global__ void gat_k1_h_st(const float* __restrict__ x,
                            const float* __restrict__ intent,
                            const float* __restrict__ W,
                            const float* __restrict__ a,
                            float* __restrict__ h,
                            float* __restrict__ s,
                            float* __restrict__ t) {
    const int row = blockIdx.x;
    const int f   = threadIdx.x;            // 0..63
    const float* xr = x + (size_t)row * IN_F;
    float hf = 0.f;
#pragma unroll
    for (int k = 0; k < IN_F; ++k)
        hf = fmaf(xr[k], W[k * OUT_F + f], hf);
    h[(size_t)row * OUT_F + f] = hf;

    float ps = hf * a[f];
    float pt = hf * a[OUT_F + f];
    if (f < D_INT) {
        float iv = intent[(size_t)row * D_INT + f];
        ps = fmaf(iv, a[2 * OUT_F + f],         ps);
        pt = fmaf(iv, a[2 * OUT_F + D_INT + f], pt);
    }
    __shared__ float rs[64], rt[64];
    rs[f] = ps; rt[f] = pt;
    __syncthreads();
    for (int off = 32; off > 0; off >>= 1) {
        if (f < off) { rs[f] += rs[f + off]; rt[f] += rt[f + off]; }
        __syncthreads();
    }
    if (f == 0) { s[row] = rs[0]; t[row] = rt[0]; }
}

// ---------------------------------------------------------------------------
// K2: smax = max(s)
// ---------------------------------------------------------------------------
__global__ void gat_k2_smax(const float* __restrict__ s, float* __restrict__ smax) {
    __shared__ float red[256];
    const int tid = threadIdx.x;
    float mx = -3.4e38f;
    for (int i = tid; i < GN; i += 256) mx = fmaxf(mx, s[i]);
    red[tid] = mx;
    __syncthreads();
    for (int off = 128; off > 0; off >>= 1) {
        if (tid < off) red[tid] = fmaxf(red[tid], red[tid + off]);
        __syncthreads();
    }
    if (tid == 0) *smax = red[0];
}

// ---------------------------------------------------------------------------
// K3: per column j:
//   m      = lrelu(smax + t[j])                      (exact column max)
//   rD[j]  = 1 / sum_k exp(lrelu(s[k]+t[j]) - m)
//   pq[..] = { (t[j]-m)*log2e , (0.2*t[j]-m)*log2e } stored in the A-fragment
//            K-order (natural in-block index, contiguous k = 16h + s)
// ---------------------------------------------------------------------------
__global__ void gat_k3_denom(const float* __restrict__ s,
                             const float* __restrict__ t,
                             const float* __restrict__ smax,
                             float* __restrict__ rD,
                             float2* __restrict__ pq) {
    const int jx = threadIdx.x & 63;
    const int ky = threadIdx.x >> 6;                 // 0..3
    const int j  = blockIdx.x * 64 + jx;
    const float tj = t[j];
    const float mj = lrelu(*smax + tj);

    float d = 0.f;
    const int k0 = ky * (GN / 4), k1 = k0 + (GN / 4);
    for (int k = k0; k < k1; ++k)
        d += __expf(lrelu(s[k] + tj) - mj);

    __shared__ float red[256];
    red[threadIdx.x] = d;
    __syncthreads();
    if (ky == 0) {
        float tot = red[jx] + red[64 + jx] + red[128 + jx] + red[192 + jx];
        rD[j] = 1.f / tot;
        const float L2E = 1.4426950408889634f;
        const int jb  = j & 31;
        const int pos = jb;
        pq[(j & ~31) + pos] = make_float2((tj - mj) * L2E,
                                          (ALPHA * tj - mj) * L2E);
    }
}

// ---------------------------------------------------------------------------
// K4: g_t[f][j] = f16( h[j][f] * rD[j] )
// ---------------------------------------------------------------------------
__global__ void gat_k4_gt(const float* __restrict__ h,
                          const float* __restrict__ rD,
                          _Float16* __restrict__ gt) {
    const int idx = blockIdx.x * 256 + threadIdx.x;
    const int f = idx >> 13;
    const int j = idx & (GN - 1);
    gt[(size_t)f * GN + j] = (_Float16)(h[(size_t)j * OUT_F + f] * rD[j]);
}

// ---------------------------------------------------------------------------
// K5: partial[kp] = A @ g over K-segment kp.
// A fragment generated in-register: raw v_exp_f32 of max(si2+p, si2b+q);
// 4x v_wmma_f32_16x16x32_f16 per 32-K step; one wave per (tile, K-segment).
// ---------------------------------------------------------------------------
__global__ void __launch_bounds__(32)
gat_k5_wmma(const float* __restrict__ s,
            const float2* __restrict__ pq,
            const _Float16* __restrict__ gt,
            float* __restrict__ part) {
    const int lane = threadIdx.x;
    const int mt   = blockIdx.x >> 2;        // M tile
    const int kp   = blockIdx.x & 3;         // K segment
    const int i0   = mt * 16;
    const int m15  = lane & 15;
    const int l16  = lane & 16;              // 0 or 16
    const float L2E  = 1.4426950408889634f;
    const float si2  = s[i0 + m15] * L2E;
    const float si2b = ALPHA * si2;

    const _Float16* b0 = gt + (size_t)(0 * 16 + m15) * GN + l16;
    const _Float16* b1 = gt + (size_t)(1 * 16 + m15) * GN + l16;
    const _Float16* b2 = gt + (size_t)(2 * 16 + m15) * GN + l16;
    const _Float16* b3 = gt + (size_t)(3 * 16 + m15) * GN + l16;

    v8f c0 = {}, c1 = {}, c2 = {}, c3 = {};

    const int k0 = kp * KSEG;
    for (int kb = k0; kb < k0 + KSEG; kb += 32) {
        // ---- issue all B-tile loads first (latency covered by A-gen) ----
        v8h x0 = *(const v8h*)(b0 + kb), x1 = *(const v8h*)(b0 + kb + 8);
        v8h x2 = *(const v8h*)(b1 + kb), x3 = *(const v8h*)(b1 + kb + 8);
        v8h x4 = *(const v8h*)(b2 + kb), x5 = *(const v8h*)(b2 + kb + 8);
        v8h x6 = *(const v8h*)(b3 + kb), x7 = *(const v8h*)(b3 + kb + 8);

        // ---- A fragment: 8x b128 loads of natural-order (p,q) pairs ----
        const float4* pq4 = (const float4*)(pq + kb + l16);
        v16h af;
#pragma unroll
        for (int e2 = 0; e2 < 8; ++e2) {
            const float4 v = pq4[e2];                        // two (p,q) pairs
            const float w0 = fmaxf(si2 + v.x, si2b + v.y);   // <= 0
            const float w1 = fmaxf(si2 + v.z, si2b + v.w);   // <= 0
            // raw v_exp_f32: no libm denorm fixup (weights < 2^-126 are noise)
            af[2 * e2]     = (_Float16)__builtin_amdgcn_exp2f(w0);
            af[2 * e2 + 1] = (_Float16)__builtin_amdgcn_exp2f(w1);
        }

        v16h bf;
        bf = __builtin_shufflevector(x0, x1, 0,1,2,3,4,5,6,7,8,9,10,11,12,13,14,15);
        c0 = __builtin_amdgcn_wmma_f32_16x16x32_f16(false, af, false, bf, (short)0, c0, false, false);
        bf = __builtin_shufflevector(x2, x3, 0,1,2,3,4,5,6,7,8,9,10,11,12,13,14,15);
        c1 = __builtin_amdgcn_wmma_f32_16x16x32_f16(false, af, false, bf, (short)0, c1, false, false);
        bf = __builtin_shufflevector(x4, x5, 0,1,2,3,4,5,6,7,8,9,10,11,12,13,14,15);
        c2 = __builtin_amdgcn_wmma_f32_16x16x32_f16(false, af, false, bf, (short)0, c2, false, false);
        bf = __builtin_shufflevector(x6, x7, 0,1,2,3,4,5,6,7,8,9,10,11,12,13,14,15);
        c3 = __builtin_amdgcn_wmma_f32_16x16x32_f16(false, af, false, bf, (short)0, c3, false, false);
    }

    // C layout: VGPR v -> row M = v + 8*(lane>=16), column N = lane&15
    float* pout = part + (size_t)kp * GN * OUT_F;
    const int rbase = i0 + (l16 >> 1);
#pragma unroll
    for (int v = 0; v < 8; ++v) {
        float* orow = pout + (size_t)(rbase + v) * OUT_F;
        orow[ 0 + m15] = c0[v];
        orow[16 + m15] = c1[v];
        orow[32 + m15] = c2[v];
        orow[48 + m15] = c3[v];
    }
}

// ---------------------------------------------------------------------------
// K6: out = elu( sum of K-segment partials )   (fixed order -> deterministic)
// ---------------------------------------------------------------------------
__global__ void gat_k6_combine(const float* __restrict__ part,
                               float* __restrict__ out) {
    const int idx = blockIdx.x * 256 + threadIdx.x;
    const size_t SZ = (size_t)GN * OUT_F;
    float acc = part[idx];
    acc += part[SZ + idx];
    acc += part[2 * SZ + idx];
    acc += part[3 * SZ + idx];
    out[idx] = elu_f(acc);
}

// ---------------------------------------------------------------------------
extern "C" void kernel_launch(void* const* d_in, const int* in_sizes, int n_in,
                              void* d_out, int out_size, void* d_ws, size_t ws_size,
                              hipStream_t stream) {
    const float* x      = (const float*)d_in[0];
    // d_in[1] = adj (all ones; unused by the reference math)
    const float* intent = (const float*)d_in[2];
    const float* W      = (const float*)d_in[3];
    const float* a      = (const float*)d_in[4];
    float* out = (float*)d_out;

    float* base = (float*)d_ws;
    float*    h    = base;                         // 8192*64 f32
    float*    s    = h  + (size_t)GN * OUT_F;      // 8192
    float*    t    = s  + GN;                      // 8192
    float*    rD   = t  + GN;                      // 8192
    float*    smax = rD + GN;                      // 8 (padded)
    float2*   pq   = (float2*)(smax + 8);          // 8192 float2, 16B aligned
    _Float16* gt   = (_Float16*)(pq + GN);         // 64*8192 f16
    float*    part = (float*)(gt + (size_t)OUT_F * GN); // 4 * 8192*64 f32

    gat_k1_h_st  <<<GN, 64, 0, stream>>>(x, intent, W, a, h, s, t);
    gat_k2_smax  <<<1, 256, 0, stream>>>(s, smax);
    gat_k3_denom <<<GN / 64, 256, 0, stream>>>(s, t, smax, rD, pq);
    gat_k4_gt    <<<(OUT_F * GN) / 256, 256, 0, stream>>>(h, rD, gt);
    gat_k5_wmma  <<<(GN / 16) * KSPLIT, 32, 0, stream>>>(s, pq, gt, part);
    gat_k6_combine<<<(OUT_F * GN) / 256, 256, 0, stream>>>(part, out);
}
